// S2VT_70909910057271
// MI455X (gfx1250) — matrix operand order repaired and checked by reading, not compiled, not run on hardware
//
#include <hip/hip_runtime.h>
#include <hip/hip_bf16.h>
#include <math.h>

// ---------------- model dimensions (fixed by reference) ----------------
#define T_ENCL 80
#define MAXLEN 80
#define BB     64
#define DIN    4096
#define EUD    1024   // encoder hidden
#define DUD    1024   // decoder hidden
#define VOCABN 20000
#define ATTN   80

// GEMM blocking
#define BN_TILES 8            // N-tiles (of 16) per 256-thread block
#define KBLK     32           // K panel per stage
#define LROW     40           // LDS row stride in bf16 elems (32 + 8 pad)
#define ABUF     (64 * LROW)  // one LDS A panel (64 rows x 32 k)

typedef unsigned short u16;
typedef __attribute__((ext_vector_type(16))) __bf16         v16bf;
typedef __attribute__((ext_vector_type(8)))  float          v8f;
typedef __attribute__((ext_vector_type(8)))  unsigned short ushort8;

// fp32 -> bf16 round-to-nearest-even
__device__ __forceinline__ u16 f2bf(float x) {
    unsigned int u = __float_as_uint(x);
    unsigned int r = 0x7FFFu + ((u >> 16) & 1u);
    return (u16)((u + r) >> 16);
}
__device__ __forceinline__ float bf2f(u16 b) {
    return __uint_as_float(((unsigned int)b) << 16);
}

union FragU { v16bf v; u16 u[16]; ushort8 h[2]; };

// ---- A fragment (16x32 bf16) from LDS, ISA 7.12.2 "16-bit A-Matrix 16x32" ----
// lanes 0-15 : row=lane,    v0..v3 = K 0..7,  v4..v7 = K 16..23
// lanes 16-31: row=lane-16, v0..v3 = K 8..15, v4..v7 = K 24..31
__device__ __forceinline__ v16bf load_a_lds(const u16* base) {
    int lane = threadIdx.x & 31;
    const u16* p = base + (lane & 15) * LROW + ((lane >> 4) & 1) * 8;
    FragU f;
    f.h[0] = *(const ushort8*)p;          // K +0..7
    f.h[1] = *(const ushort8*)(p + 16);   // K +16..23
    return f.v;
}
// ---- B fragment (32x16 bf16): lane = N col; lanes 0-15: K 0..7,16..23, lanes 16-31: K 8..15,24..31
// W stored [N,K] row-major -> two b128 loads per lane.
__device__ __forceinline__ v16bf load_b_bf(const u16* __restrict__ base, long ldb) {
    int lane = threadIdx.x & 31;
    const u16* p = base + (long)(lane & 15) * ldb + ((lane >> 4) & 1) * 8;
    FragU f;
    f.h[0] = *(const ushort8*)p;
    f.h[1] = *(const ushort8*)(p + 16);
    return f.v;
}

// One operand's K accumulation, software pipelined:
//   - A panels double-buffered in LDS; next panel's global load is issued before
//     the WMMA block and its ds_store sunk after it (hides global latency),
//   - next B fragment loaded during current compute,
//   - all 4 A fragments loaded before the 4 WMMAs (one ds clause).
__device__ __forceinline__ void op_accum(v8f (&acc)[4], u16* sA,
                                         const u16* __restrict__ A, long lda,
                                         const u16* __restrict__ W, long ldw, int K,
                                         long m0, long n0, bool active)
{
    const int tid = threadIdx.x;
    const int row = tid >> 2;        // 0..63
    const int ch  = tid & 3;         // 4 x 8-elem (16B) chunks per 32-elem row
    const u16* __restrict__ arow = A + (m0 + row) * lda + ch * 8;
    const u16* __restrict__ brow = W + n0 * ldw;

    // prologue: stage panel 0, preload B fragment 0
    *(ushort8*)(sA + row * LROW + ch * 8) = *(const ushort8*)arow;
    __syncthreads();
    v16bf bcur = {};
    if (active) bcur = load_b_bf(brow, ldw);

    int p = 0;
    for (int k = 0; k < K; k += KBLK) {
        const bool has_next = (k + KBLK) < K;
        ushort8 stg;
        if (has_next)                               // issue next A panel load now...
            stg = *(const ushort8*)(arow + k + KBLK);
        v16bf bnext = bcur;
        if (active && has_next) {                   // ...and next B fragment
            bnext = load_b_bf(brow + k + KBLK, ldw);
            __builtin_prefetch((const void*)(brow + (long)(tid & 15) * ldw + k + 2 * KBLK + 256), 0, 1);
        }
        if (active) {
            const u16* ap = sA + p * ABUF;
            v16bf a0 = load_a_lds(ap);
            v16bf a1 = load_a_lds(ap + 16 * LROW);
            v16bf a2 = load_a_lds(ap + 32 * LROW);
            v16bf a3 = load_a_lds(ap + 48 * LROW);
            acc[0] = __builtin_amdgcn_wmma_f32_16x16x32_bf16(false, a0, false, bcur, (short)0, acc[0], false, false);
            acc[1] = __builtin_amdgcn_wmma_f32_16x16x32_bf16(false, a1, false, bcur, (short)0, acc[1], false, false);
            acc[2] = __builtin_amdgcn_wmma_f32_16x16x32_bf16(false, a2, false, bcur, (short)0, acc[2], false, false);
            acc[3] = __builtin_amdgcn_wmma_f32_16x16x32_bf16(false, a3, false, bcur, (short)0, acc[3], false, false);
        }
        if (has_next)                               // ds_store sunk below the WMMAs
            *(ushort8*)(sA + (1 - p) * ABUF + row * LROW + ch * 8) = stg;
        __syncthreads();
        bcur = bnext;
        p ^= 1;
    }
}

// ---------------------------------------------------------------------------
// Block GEMM: 256 threads = 8 waves; block computes a 64 x 128 tile of C.
// C[M,N] = sum_i A_i[M,K_i] * W_i[N,K_i]^T (+biases) -> PReLU? -> +addmat?
// A_i, W_i bf16; C fp32 and/or bf16.  M % 64 == 0, K_i % 32 == 0, N % 16 == 0.
// ---------------------------------------------------------------------------
__global__ __launch_bounds__(256)
void wmma_gemm_lds(const u16* __restrict__ A0, long lda0, const u16* __restrict__ W0, long ldw0, int K0,
                   const u16* __restrict__ A1, long lda1, const u16* __restrict__ W1, long ldw1, int K1,
                   const u16* __restrict__ A2, long lda2, const u16* __restrict__ W2, long ldw2, int K2,
                   const float* __restrict__ bias0, const float* __restrict__ bias1,
                   const float* __restrict__ addmat, long ldadd,
                   const float* __restrict__ prelu,
                   float* __restrict__ C, u16* __restrict__ Cb,
                   long ldc, int M, int N)
{
    __shared__ __align__(16) u16 sA[2 * ABUF];

    int nT    = N >> 4;
    int nBlkN = (nT + BN_TILES - 1) / BN_TILES;
    int bm    = blockIdx.x / nBlkN;
    int bn    = blockIdx.x - bm * nBlkN;
    long m0   = (long)bm << 6;
    int wave  = threadIdx.x >> 5;
    int ntile = bn * BN_TILES + wave;
    bool active = ntile < nT;            // wave-uniform; inactive waves still barrier
    long n0   = (long)(active ? ntile : 0) << 4;

    v8f acc[4] = {};
    op_accum(acc, sA, A0, lda0, W0, ldw0, K0, m0, n0, active);
    if (A1) op_accum(acc, sA, A1, lda1, W1, ldw1, K1, m0, n0, active);
    if (A2) op_accum(acc, sA, A2, lda2, W2, ldw2, K2, m0, n0, active);
    if (!active) return;

    // C/D layout: lanes 0-15: N=lane, rows +0..7; lanes 16-31: N=lane-16, rows +8..15
    int  lane = threadIdx.x & 31;
    int  col  = lane & 15;
    int  rb   = (lane >> 4) * 8;
    long n    = n0 + col;
    float bsum = 0.f;
    if (bias0) bsum += bias0[n];
    if (bias1) bsum += bias1[n];
    float slope = prelu ? prelu[0] : 0.f;
#pragma unroll
    for (int j = 0; j < 4; ++j) {
#pragma unroll
        for (int r = 0; r < 8; ++r) {
            long  m = m0 + 16 * j + rb + r;
            float v = acc[j][r] + bsum;
            if (prelu)  v = (v >= 0.f) ? v : slope * v;   // PReLU before residual add
            if (addmat) v += addmat[m * ldadd + n];
            if (C)  C[m * ldc + n]  = v;
            if (Cb) Cb[m * ldc + n] = f2bf(v);
        }
    }
}

// ---------------- elementwise LSTM cell (gate order i,f,g,o) ----------------
__global__ void lstm_cell_kernel(const float* __restrict__ g,
                                 float* __restrict__ c,
                                 u16* __restrict__ hb,
                                 u16* __restrict__ seq_b,
                                 float* __restrict__ seq_f, int H)
{
    int idx = blockIdx.x * blockDim.x + threadIdx.x;
    if (idx >= BB * H) return;
    int b = idx / H, j = idx - b * H;
    const float* gr = g + (long)b * 4 * H;
    float gi = gr[j], gf = gr[H + j], gg = gr[2 * H + j], go = gr[3 * H + j];
    float si = 1.f / (1.f + expf(-gi));
    float sf = 1.f / (1.f + expf(-gf));
    float so = 1.f / (1.f + expf(-go));
    float cn = sf * c[idx] + si * tanhf(gg);
    c[idx] = cn;
    float hn = so * tanhf(cn);
    u16 hbf = f2bf(hn);
    hb[idx] = hbf;
    if (seq_b) seq_b[idx] = hbf;
    if (seq_f) seq_f[idx] = hn;
}

// softmax over 80 slots, 64 rows (fp32)
__global__ void softmax80_kernel(float* __restrict__ w)
{
    int b = blockIdx.x * blockDim.x + threadIdx.x;
    if (b >= BB) return;
    float* row = w + (long)b * ATTN;
    float mx = row[0];
    for (int s = 1; s < ATTN; ++s) mx = fmaxf(mx, row[s]);
    float sum = 0.f;
    for (int s = 0; s < ATTN; ++s) { float e = expf(row[s] - mx); row[s] = e; sum += e; }
    float inv = 1.f / sum;
    for (int s = 0; s < ATTN; ++s) row[s] *= inv;
}

// f[b,h] = sum_s w[b,s] * E[s,b,h]; E bf16, f emitted bf16
__global__ void attn_wsum_kernel(const float* __restrict__ w,
                                 const u16* __restrict__ Eb,
                                 u16* __restrict__ fb)
{
    int idx = blockIdx.x * blockDim.x + threadIdx.x;   // b*EU + h
    if (idx >= BB * EUD) return;
    int b = idx / EUD;
    const float* wr = w + (long)b * ATTN;
    float acc = 0.f;
#pragma unroll 4
    for (int s = 0; s < T_ENCL; ++s)
        acc += wr[s] * bf2f(Eb[(long)s * BB * EUD + idx]);
    fb[idx] = f2bf(acc);
}

// samples[0] = in_emb_W[:, VOCAB-2] + in_emb_b (BOS one-hot), bf16 out
__global__ void sample0_kernel(const float* __restrict__ Wemb,
                               const float* __restrict__ bemb,
                               u16* __restrict__ out)
{
    int idx = blockIdx.x * blockDim.x + threadIdx.x;   // b*DU + j
    if (idx >= BB * DUD) return;
    int j = idx % DUD;
    out[idx] = f2bf(Wemb[(long)j * VOCABN + (VOCABN - 2)] + bemb[j]);
}

__global__ void cvt_bf16_kernel(const float* __restrict__ s, u16* __restrict__ d, long n)
{
    long i = (long)blockIdx.x * blockDim.x + threadIdx.x;
    if (i < n) d[i] = f2bf(s[i]);
}

__global__ void zero_u32_kernel(unsigned int* __restrict__ p, long nwords)
{
    long i = (long)blockIdx.x * blockDim.x + threadIdx.x;
    if (i < nwords) p[i] = 0u;
}

// ---------------------------------------------------------------------------
extern "C" void kernel_launch(void* const* d_in, const int* in_sizes, int n_in,
                              void* d_out, int out_size, void* d_ws, size_t ws_size,
                              hipStream_t stream)
{
    const float* input_data = (const float*)d_in[0];
    const float* correct    = (const float*)d_in[1];
    const float* eWih = (const float*)d_in[2];
    const float* eWhh = (const float*)d_in[3];
    const float* eBih = (const float*)d_in[4];
    const float* eBhh = (const float*)d_in[5];
    const float* dWih = (const float*)d_in[6];
    const float* dWhh = (const float*)d_in[7];
    const float* dBih = (const float*)d_in[8];
    const float* dBhh = (const float*)d_in[9];
    const float* inW  = (const float*)d_in[10];
    const float* inB  = (const float*)d_in[11];
    const float* oW   = (const float*)d_in[12];
    const float* oB   = (const float*)d_in[13];
    const float* aW1  = (const float*)d_in[14];
    const float* aB1  = (const float*)d_in[15];
    const float* aW2  = (const float*)d_in[16];
    const float* aB2  = (const float*)d_in[17];
    const float* prelu = (const float*)d_in[18];
    float* out = (float*)d_out;

    // ---- workspace carve-out (256B aligned blocks) ----
    char* wsp = (char*)d_ws;
    auto alloc = [&](size_t bytes) -> void* {
        void* p = (void*)wsp; wsp += (bytes + 255) & ~(size_t)255; return p;
    };
    // fp32 buffers
    float* enc_pad_seq = (float*)alloc((size_t)MAXLEN * BB * EUD * 4);
    float* gbuf        = (float*)alloc((size_t)BB * 4 * DUD * 4);
    float* attw        = (float*)alloc((size_t)BB * ATTN * 4);
    float* enc_c       = (float*)alloc((size_t)BB * EUD * 4);
    float* dec_c       = (float*)alloc((size_t)BB * DUD * 4);
    // bf16 activation buffers
    u16* enc_hb     = (u16*)alloc((size_t)BB * EUD * 2);
    u16* dec_hb     = (u16*)alloc((size_t)BB * DUD * 2);
    u16* fbuf_b     = (u16*)alloc((size_t)BB * EUD * 2);
    u16* ctx_b      = (u16*)alloc((size_t)BB * DUD * 2);
    u16* enc_seq_b  = (u16*)alloc((size_t)T_ENCL * BB * EUD * 2);
    u16* samples_b  = (u16*)alloc((size_t)MAXLEN * BB * DUD * 2);
    u16* xb_input   = (u16*)alloc((size_t)T_ENCL * BB * DIN * 2);
    u16* correct_b  = (u16*)alloc((size_t)(MAXLEN - 1) * BB * VOCABN * 2);
    // bf16 weight buffers
    u16* wb_eWih = (u16*)alloc((size_t)4 * EUD * DIN * 2);
    u16* wb_eWhh = (u16*)alloc((size_t)4 * EUD * EUD * 2);
    u16* wb_dWih = (u16*)alloc((size_t)4 * DUD * 2 * DUD * 2);
    u16* wb_dWhh = (u16*)alloc((size_t)4 * DUD * DUD * 2);
    u16* wb_inW  = (u16*)alloc((size_t)DUD * VOCABN * 2);
    u16* wb_oW   = (u16*)alloc((size_t)VOCABN * DUD * 2);
    u16* wb_aW1  = (u16*)alloc((size_t)ATTN * 2 * DUD * 2);
    u16* wb_aW2  = (u16*)alloc((size_t)DUD * 2 * DUD * 2);

    auto cvt = [&](const float* s, u16* d, long n) {
        cvt_bf16_kernel<<<(int)((n + 255) / 256), 256, 0, stream>>>(s, d, n);
    };
    auto zero = [&](void* p, long nwords) {
        zero_u32_kernel<<<(int)((nwords + 255) / 256), 256, 0, stream>>>((unsigned int*)p, nwords);
    };
    auto gemm = [&](const u16* A0, long lda0, const u16* W0, long ldw0, int K0,
                    const u16* A1, long lda1, const u16* W1, long ldw1, int K1,
                    const u16* A2, long lda2, const u16* W2, long ldw2, int K2,
                    const float* b0, const float* b1,
                    const float* add, long ldadd, const float* pre,
                    float* C, u16* Cb, long ldc, int M, int N) {
        int nT    = N / 16;
        int nBlkN = (nT + BN_TILES - 1) / BN_TILES;
        int grid  = (M / 64) * nBlkN;
        wmma_gemm_lds<<<grid, 256, 0, stream>>>(A0, lda0, W0, ldw0, K0,
                                                A1, lda1, W1, ldw1, K1,
                                                A2, lda2, W2, ldw2, K2,
                                                b0, b1, add, ldadd, pre, C, Cb, ldc, M, N);
    };

    // ---- Phase 0: one-time conversions + state zeroing ----
    cvt(eWih, wb_eWih, (long)4 * EUD * DIN);
    cvt(eWhh, wb_eWhh, (long)4 * EUD * EUD);
    cvt(dWih, wb_dWih, (long)4 * DUD * 2 * DUD);
    cvt(dWhh, wb_dWhh, (long)4 * DUD * DUD);
    cvt(inW,  wb_inW,  (long)DUD * VOCABN);
    cvt(oW,   wb_oW,   (long)VOCABN * DUD);
    cvt(aW1,  wb_aW1,  (long)ATTN * 2 * DUD);
    cvt(aW2,  wb_aW2,  (long)DUD * 2 * DUD);
    cvt(input_data, xb_input, (long)T_ENCL * BB * DIN);
    cvt(correct + (size_t)BB * VOCABN, correct_b, (long)(MAXLEN - 1) * BB * VOCABN);
    zero(enc_c, BB * EUD);
    zero(dec_c, BB * DUD);
    zero(enc_hb, BB * EUD / 2);
    zero(dec_hb, BB * DUD / 2);

    // ---- Phase 1: encoder over input_data ----
    for (int t = 0; t < T_ENCL; ++t) {
        gemm(xb_input + (size_t)t * BB * DIN, DIN, wb_eWih, DIN, DIN,
             enc_hb, EUD, wb_eWhh, EUD, EUD,
             nullptr, 0, nullptr, 0, 0,
             eBih, eBhh, nullptr, 0, nullptr,
             gbuf, nullptr, 4 * EUD, BB, 4 * EUD);
        lstm_cell_kernel<<<(BB * EUD + 255) / 256, 256, 0, stream>>>(
            gbuf, enc_c, enc_hb, enc_seq_b + (size_t)t * BB * EUD, nullptr, EUD);
    }

    // ---- Phase 2: decoder prime over [enc_seq, 0] ----
    for (int t = 0; t < T_ENCL; ++t) {
        gemm(enc_seq_b + (size_t)t * BB * EUD, EUD, wb_dWih, 2 * DUD, EUD,
             dec_hb, DUD, wb_dWhh, DUD, DUD,
             nullptr, 0, nullptr, 0, 0,
             dBih, dBhh, nullptr, 0, nullptr,
             gbuf, nullptr, 4 * DUD, BB, 4 * DUD);
        lstm_cell_kernel<<<(BB * DUD + 255) / 256, 256, 0, stream>>>(
            gbuf, dec_c, dec_hb, nullptr, nullptr, DUD);
    }

    // ---- Phase 3: encoder over zero padding (continues encoder state) ----
    for (int t = 0; t < MAXLEN; ++t) {
        gemm(enc_hb, EUD, wb_eWhh, EUD, EUD,
             nullptr, 0, nullptr, 0, 0,
             nullptr, 0, nullptr, 0, 0,
             eBih, eBhh, nullptr, 0, nullptr,
             gbuf, nullptr, 4 * EUD, BB, 4 * EUD);
        lstm_cell_kernel<<<(BB * EUD + 255) / 256, 256, 0, stream>>>(
            gbuf, enc_c, enc_hb, nullptr, enc_pad_seq + (size_t)t * BB * EUD, EUD);
    }

    // ---- Phase 4: teacher-forced samples ----
    sample0_kernel<<<(BB * DUD + 255) / 256, 256, 0, stream>>>(inW, inB, samples_b);
    gemm(correct_b, VOCABN, wb_inW, VOCABN, VOCABN,
         nullptr, 0, nullptr, 0, 0,
         nullptr, 0, nullptr, 0, 0,
         inB, nullptr, nullptr, 0, nullptr,
         nullptr, samples_b + (size_t)BB * DUD, DUD, (MAXLEN - 1) * BB, DUD);

    // ---- Phase 5: attention decoder loop ----
    for (int t = 0; t < MAXLEN; ++t) {
        const u16* samp = samples_b + (size_t)t * BB * DUD;
        // logits = [h, samp] @ att_W1^T + b1                [64, 80]
        gemm(dec_hb, DUD, wb_aW1, 2 * DUD, DUD,
             samp, DUD, wb_aW1 + DUD, 2 * DUD, DUD,
             nullptr, 0, nullptr, 0, 0,
             aB1, nullptr, nullptr, 0, nullptr,
             attw, nullptr, ATTN, BB, ATTN);
        softmax80_kernel<<<1, 64, 0, stream>>>(attw);
        attn_wsum_kernel<<<(BB * EUD + 255) / 256, 256, 0, stream>>>(attw, enc_seq_b, fbuf_b);
        // context = PReLU([samp, f] @ att_W2^T + b2) + enc_pad_seq[t]  (bf16 out)
        gemm(samp, DUD, wb_aW2, 2 * DUD, DUD,
             fbuf_b, EUD, wb_aW2 + DUD, 2 * DUD, EUD,
             nullptr, 0, nullptr, 0, 0,
             aB2, nullptr, enc_pad_seq + (size_t)t * BB * EUD, EUD, prelu,
             nullptr, ctx_b, DUD, BB, DUD);
        // g = [samp, ctx] @ dec_W_ih^T + h @ dec_W_hh^T + biases
        gemm(samp, DUD, wb_dWih, 2 * DUD, DUD,
             ctx_b, DUD, wb_dWih + DUD, 2 * DUD, DUD,
             dec_hb, DUD, wb_dWhh, DUD, DUD,
             dBih, dBhh, nullptr, 0, nullptr,
             gbuf, nullptr, 4 * DUD, BB, 4 * DUD);
        lstm_cell_kernel<<<(BB * DUD + 255) / 256, 256, 0, stream>>>(
            gbuf, dec_c, dec_hb, nullptr, nullptr, DUD);
        // word = h @ out_emb_W^T + b                        [64, 20000]
        gemm(dec_hb, DUD, wb_oW, DUD, DUD,
             nullptr, 0, nullptr, 0, 0,
             nullptr, 0, nullptr, 0, 0,
             oB, nullptr, nullptr, 0, nullptr,
             out + (size_t)t * BB * VOCABN, nullptr, VOCABN, BB, VOCABN);
    }

    (void)in_sizes; (void)n_in; (void)out_size; (void)ws_size;
}
